// Encoder_42262478192787
// MI455X (gfx1250) — compile-verified, matrix operand order checked by reading, not run on hardware
//
#include <hip/hip_runtime.h>
#include <hip/hip_bf16.h>
#include <math.h>

// ---------------------------------------------------------------------------
// CDNA5 (gfx1250) implementation: multi-view embedding -> MoE (dense experts,
// top-4 gate weights) -> fractional positional encoding -> 3-layer post-norm
// transformer. Large matmuls: v_wmma_f32_16x16x32_f16 (f32 accumulate).
// Expert phase-2 weight panels are streamed with the Tensor Data Mover
// (tensor_load_to_lds + s_wait_tensorcnt, double-buffered in LDS).
// ---------------------------------------------------------------------------

typedef __attribute__((ext_vector_type(16))) _Float16 v16h;
typedef __attribute__((ext_vector_type(8)))  float    v8f;
typedef __attribute__((ext_vector_type(4)))  unsigned int v4u;
typedef __attribute__((ext_vector_type(8)))  int      v8i;
typedef __attribute__((ext_vector_type(4)))  int      v4i;

#define T_TOK 288      // B*L = 32*9
#define D_MODEL 512
#define N_EXP 16
#define N_VIEW 3

#if defined(__has_builtin)
#if __has_builtin(__builtin_amdgcn_tensor_load_to_lds) && \
    __has_builtin(__builtin_amdgcn_s_wait_tensorcnt) &&    \
    __has_builtin(__builtin_amdgcn_readfirstlane)
#define HAVE_TDM 1
#else
#define HAVE_TDM 0
#endif
#else
#define HAVE_TDM 0
#endif

__device__ __forceinline__ v8f wmma16x16x32(v16h a, v16h b, v8f c) {
  // (neg_a, A, neg_b, B, c_mod, C, reuse_a, reuse_b)
  return __builtin_amdgcn_wmma_f32_16x16x32_f16(false, a, false, b, (short)0, c,
                                                false, false);
}

#if HAVE_TDM
// LDS byte offset of a generic pointer into shared memory.
__device__ __forceinline__ unsigned lds_offset(void* p) {
  return (unsigned)(unsigned long long)
      (__attribute__((address_space(3))) void*)p;
}

// Issue one TDM 2D tile load: W2 panel [32 rows x 16 cols] of f32 from a
// row-major [2048 x 512] tensor into LDS at lds_byte_addr.
__device__ __forceinline__ void tdm_load_panel(const float* gsrc,
                                               unsigned lds_byte_addr) {
  const unsigned long long ga = (unsigned long long)gsrc;
  v4u g0;
  g0[0] = 1u;                                          // count=1, user D#
  g0[1] = lds_byte_addr;                               // lds_addr
  g0[2] = (unsigned)(ga & 0xFFFFFFFFull);              // global_addr lo
  g0[3] = (unsigned)((ga >> 32) & 0x1FFFFFFull)        // global_addr[56:32]
        | (2u << 30);                                  // type = 2 ("image")
  v8i g1;
  g1[0] = (int)(2u << 16);       // data_size = 4B; no multicast / flags
  g1[1] = (int)(512u << 16);     // tensor_dim0 = 512 (bits 63:48)
  g1[2] = (int)(2048u << 16);    // tensor_dim1 = 2048 (bits 95:80)
  g1[3] = (int)(16u << 16);      // tile_dim0 = 16 (bits 127:112)
  g1[4] = 32;                    // tile_dim1 = 32, tile_dim2 = 0
  g1[5] = 512;                   // tensor_dim0_stride = 512 (lo32)
  g1[6] = 0;                     // stride0 hi / stride1 lo
  g1[7] = 0;                     // stride1 hi
  const v4i gz = {0, 0, 0, 0};
#if __clang_major__ >= 23
  const v8i gz8 = {0, 0, 0, 0, 0, 0, 0, 0};
  __builtin_amdgcn_tensor_load_to_lds(g0, g1, gz, gz, gz8, 0);
#else
  __builtin_amdgcn_tensor_load_to_lds(g0, g1, gz, gz, 0);
#endif
}
#endif  // HAVE_TDM

// ---------------------------------------------------------------------------
// Generic WMMA GEMM: C[M x N] = act(A[M x K] @ B[K x N] + bias)
// One wave per 16x16 C tile. M, N multiples of 16; K arbitrary (masked tail).
// GATHER: A row t is A[gidx[t]] (embedding lookup).  ACT: 0 none, 2 relu.
// ---------------------------------------------------------------------------
template<int ACT, bool GATHER>
__global__ __launch_bounds__(256) void gemm_wmma_kernel(
    const float* __restrict__ A, const float* __restrict__ B,
    const float* __restrict__ bias, float* __restrict__ C,
    const int* __restrict__ gidx,
    int M, int N, int K, int lda, int ldb, int ldc)
{
  const int lane = threadIdx.x & 31;
  const int wave = threadIdx.x >> 5;
  const int ntiles = N >> 4;
  const int gid = blockIdx.x * (blockDim.x >> 5) + wave;
  const int mt = gid / ntiles;
  const int nt = gid - mt * ntiles;
  if (mt * 16 >= M) return;                 // wave-uniform
  const int m0 = mt * 16, n0 = nt * 16;
  const int r    = lane & 15;
  const int koff = (lane >> 4) * 8;
  const int arow = m0 + r;
  const float* __restrict__ Arow =
      A + (size_t)(GATHER ? gidx[arow] : arow) * (size_t)lda;
  const int bcol = n0 + (lane & 15);

  v8f acc = {0.f,0.f,0.f,0.f,0.f,0.f,0.f,0.f};
  for (int k0 = 0; k0 < K; k0 += 32) {
    if (k0 + 32 < K) {
      __builtin_prefetch(Arow + k0 + 32, 0, 0);
      __builtin_prefetch(&B[(size_t)(k0 + 32) * ldb + bcol], 0, 0);
    }
    v16h af, bf;
#pragma unroll
    for (int j = 0; j < 8; ++j) {
      const int k1 = k0 + koff + j;
      const int k2 = k0 + 16 + koff + j;
      af[j]     = (k1 < K) ? (_Float16)Arow[k1] : (_Float16)0.f;
      af[8 + j] = (k2 < K) ? (_Float16)Arow[k2] : (_Float16)0.f;
      float b1 = (k1 < K) ? B[(size_t)k1 * ldb + bcol] : 0.f;
      float b2 = (k2 < K) ? B[(size_t)k2 * ldb + bcol] : 0.f;
      bf[j]     = (_Float16)b1;
      bf[8 + j] = (_Float16)b2;
    }
    acc = wmma16x16x32(af, bf, acc);
  }

#pragma unroll
  for (int i = 0; i < 8; ++i) {
    const int m = m0 + koff + i;            // koff == (lane>>4)*8
    float v = acc[i] + (bias ? bias[bcol] : 0.f);
    if (ACT == 2) v = fmaxf(v, 0.f);
    C[(size_t)m * ldc + bcol] = v;
  }
}

// ---------------------------------------------------------------------------
// Laplace gating: logits = -(|v|^2 + |k_e|^2 - 2 v.k_e) + v@Wr + br,
// then top-4 softmax scattered to dense per-expert weights.
// ---------------------------------------------------------------------------
__global__ __launch_bounds__(256) void gating_kernel(
    const float* __restrict__ views,    // [3][288][512]
    const float* __restrict__ keys,     // [16][512]
    const float* __restrict__ router_w, // [3][512][16]
    const float* __restrict__ router_b, // [3][16]
    float* __restrict__ w_full)         // [3][288][16]
{
  const int vi = blockIdx.x / T_TOK;
  const int t  = blockIdx.x - vi * T_TOK;
  const float* __restrict__ v = views + ((size_t)vi * T_TOK + t) * D_MODEL;
  __shared__ float red[256];
  __shared__ float logits[N_EXP];
  __shared__ float vsq_s;
  const int tid = threadIdx.x;

  float s = 0.f;
  for (int d = tid; d < D_MODEL; d += 256) { const float x = v[d]; s += x * x; }
  red[tid] = s; __syncthreads();
  for (int off = 128; off > 0; off >>= 1) {
    if (tid < off) red[tid] += red[tid + off];
    __syncthreads();
  }
  if (tid == 0) vsq_s = red[0];
  __syncthreads();

  const int e = tid >> 4, l16 = tid & 15;   // 16 groups of 16 lanes
  float dot = 0.f, ksq = 0.f, rout = 0.f;
  for (int d = l16; d < D_MODEL; d += 16) {
    const float vd = v[d];
    const float kd = keys[e * D_MODEL + d];
    dot  += vd * kd;
    ksq  += kd * kd;
    rout += vd * router_w[((size_t)vi * D_MODEL + d) * N_EXP + e];
  }
#pragma unroll
  for (int m = 8; m > 0; m >>= 1) {
    dot  += __shfl_xor(dot,  m, 16);
    ksq  += __shfl_xor(ksq,  m, 16);
    rout += __shfl_xor(rout, m, 16);
  }
  if (l16 == 0)
    logits[e] = -(vsq_s + ksq - 2.f * dot) + rout + router_b[vi * N_EXP + e];
  __syncthreads();

  if (tid == 0) {
    float lg[N_EXP];
    bool used[N_EXP];
#pragma unroll
    for (int i = 0; i < N_EXP; ++i) { lg[i] = logits[i]; used[i] = false; }
    int   idx[4]; float val[4];
    for (int k = 0; k < 4; ++k) {
      int bi = 0; float bv = -3.4e38f;
      for (int i = 0; i < N_EXP; ++i)
        if (!used[i] && lg[i] > bv) { bv = lg[i]; bi = i; }
      used[bi] = true; idx[k] = bi; val[k] = bv;
    }
    const float mx = val[0];
    float ex[4], se = 0.f;
    for (int k = 0; k < 4; ++k) { ex[k] = expf(val[k] - mx); se += ex[k]; }
    float* wout = w_full + ((size_t)vi * T_TOK + t) * N_EXP;
    for (int i = 0; i < N_EXP; ++i) wout[i] = 0.f;
    for (int k = 0; k < 4; ++k) wout[idx[k]] = ex[k] / se;
  }
}

__global__ void zero_kernel(float* __restrict__ p, int n) {
  const int i = blockIdx.x * blockDim.x + threadIdx.x;
  if (i < n) p[i] = 0.f;
}

// ---------------------------------------------------------------------------
// Fused dense-expert MLP for one (token-tile, expert, view):
//   phase 1: H = gelu(V[16x512] @ w1_e + b1) -> LDS (f16, 64KB)
//   phase 2: Y = H @ w2_e + b2, B panels TDM-streamed into LDS (double buffer),
//            fused[t] += gate_w[t,e] * Y[t] via global f32 atomics.
// 512 threads = 16 waves.
// ---------------------------------------------------------------------------
__global__ __launch_bounds__(512) void expert_kernel(
    const float* __restrict__ views,  // [3][288][512]
    const float* __restrict__ w_full, // [3][288][16]
    const float* __restrict__ exp_w1, // [16][512][2048]
    const float* __restrict__ exp_b1, // [16][2048]
    const float* __restrict__ exp_w2, // [16][2048][512]
    const float* __restrict__ exp_b2, // [16][512]
    float* __restrict__ fused)        // [288][512]
{
  __shared__ _Float16 Hs[16][2048];          // 64 KB
#if HAVE_TDM
  __shared__ float Ws[16][2][32][16];        // 64 KB TDM staging, per wave x2
#endif

  const int tt = blockIdx.x, e = blockIdx.y, vi = blockIdx.z;
  const int m0 = tt * 16;

  // early-out if no token in this tile routed to expert e (weights >= 0)
  float wsum = 0.f;
#pragma unroll
  for (int i = 0; i < 16; ++i)
    wsum += w_full[((size_t)vi * T_TOK + m0 + i) * N_EXP + e];
  if (wsum == 0.f) return;            // block-uniform

  const int lane = threadIdx.x & 31;
  const int wave = threadIdx.x >> 5;  // 0..15
  const int r    = lane & 15;
  const int koff = (lane >> 4) * 8;

  const float* __restrict__ Arow =
      views + (((size_t)vi * T_TOK + m0 + r) * D_MODEL);
  const float* __restrict__ W1 = exp_w1 + (size_t)e * D_MODEL * 2048;
  const float* __restrict__ B1 = exp_b1 + (size_t)e * 2048;

  // ---- phase 1: H = gelu(V @ w1 + b1) -> LDS ----
  for (int t8 = 0; t8 < 8; ++t8) {
    const int n0 = (wave * 8 + t8) * 16;
    const int c  = n0 + (lane & 15);
    v8f acc = {0.f,0.f,0.f,0.f,0.f,0.f,0.f,0.f};
    for (int k0 = 0; k0 < D_MODEL; k0 += 32) {
      v16h af, bf;
#pragma unroll
      for (int j = 0; j < 8; ++j) {
        const int k1 = k0 + koff + j;
        const int k2 = k0 + 16 + koff + j;
        af[j]     = (_Float16)Arow[k1];
        af[8 + j] = (_Float16)Arow[k2];
        bf[j]     = (_Float16)W1[(size_t)k1 * 2048 + c];
        bf[8 + j] = (_Float16)W1[(size_t)k2 * 2048 + c];
      }
      acc = wmma16x16x32(af, bf, acc);
    }
#pragma unroll
    for (int i = 0; i < 8; ++i) {
      float h = acc[i] + B1[c];
      h = 0.5f * h * (1.f + erff(h * 0.70710678118f));   // exact gelu
      Hs[koff + i][c] = (_Float16)h;
    }
  }
  __syncthreads();

  // ---- phase 2: Y = H @ w2 + b2 ----
  const float* __restrict__ W2 = exp_w2 + (size_t)e * 2048 * D_MODEL;
  const float* __restrict__ B2 = exp_b2 + (size_t)e * D_MODEL;

#if HAVE_TDM
  const int wid = __builtin_amdgcn_readfirstlane(wave);
  for (int t2 = 0; t2 < 2; ++t2) {
    const int n0 = (wave * 2 + t2) * 16;
    const int c  = n0 + (lane & 15);
    const int n0u = __builtin_amdgcn_readfirstlane(n0);
    const unsigned lds_buf[2] = { lds_offset(&Ws[wid][0][0][0]),
                                  lds_offset(&Ws[wid][1][0][0]) };
    v8f acc = {0.f,0.f,0.f,0.f,0.f,0.f,0.f,0.f};

    auto compute = [&](int p, const float (*buf)[16]) {
      const int k0 = p * 32;
      v16h af, bf;
#pragma unroll
      for (int j = 0; j < 8; ++j) {
        af[j]     = Hs[r][k0 + koff + j];
        af[8 + j] = Hs[r][k0 + 16 + koff + j];
        bf[j]     = (_Float16)buf[koff + j][lane & 15];
        bf[8 + j] = (_Float16)buf[16 + koff + j][lane & 15];
      }
      acc = wmma16x16x32(af, bf, acc);
    };

    // prologue: panel 0 into buffer 0
    tdm_load_panel(W2 + (size_t)n0u, lds_buf[0]);
    for (int p = 0; p < 63; ++p) {
      // issue panel p+1 into the other buffer, then wait for panel p
      tdm_load_panel(W2 + (size_t)(p + 1) * 32 * D_MODEL + n0u,
                     lds_buf[(p + 1) & 1]);
      __builtin_amdgcn_s_wait_tensorcnt(1);
      asm volatile("" ::: "memory");
      compute(p, Ws[wid][p & 1]);
    }
    __builtin_amdgcn_s_wait_tensorcnt(0);
    asm volatile("" ::: "memory");
    compute(63, Ws[wid][1]);

#pragma unroll
    for (int i = 0; i < 8; ++i) {
      const int m = m0 + koff + i;
      const float g = w_full[((size_t)vi * T_TOK + m) * N_EXP + e];
      const float y = acc[i] + B2[c];
      atomicAdd(&fused[(size_t)m * D_MODEL + c], g * y);
    }
  }
#else
  for (int t2 = 0; t2 < 2; ++t2) {
    const int n0 = (wave * 2 + t2) * 16;
    const int c  = n0 + (lane & 15);
    v8f acc = {0.f,0.f,0.f,0.f,0.f,0.f,0.f,0.f};
    for (int k0 = 0; k0 < 2048; k0 += 32) {
      v16h af, bf;
#pragma unroll
      for (int j = 0; j < 8; ++j) {
        const int k1 = k0 + koff + j;
        const int k2 = k0 + 16 + koff + j;
        af[j]     = Hs[r][k1];
        af[8 + j] = Hs[r][k2];
        bf[j]     = (_Float16)W2[(size_t)k1 * D_MODEL + c];
        bf[8 + j] = (_Float16)W2[(size_t)k2 * D_MODEL + c];
      }
      acc = wmma16x16x32(af, bf, acc);
    }
#pragma unroll
    for (int i = 0; i < 8; ++i) {
      const int m = m0 + koff + i;
      const float g = w_full[((size_t)vi * T_TOK + m) * N_EXP + e];
      const float y = acc[i] + B2[c];
      atomicAdd(&fused[(size_t)m * D_MODEL + c], g * y);
    }
  }
#endif
}

// ---------------------------------------------------------------------------
// Fractional positional encoding (table recomputed analytically) + add.
// ---------------------------------------------------------------------------
__global__ __launch_bounds__(512) void pe_add_kernel(
    const float* __restrict__ fused, const float* __restrict__ frac,
    float* __restrict__ x)
{
  const int t = blockIdx.x, d = threadIdx.x;
  const float r = frac[t];
  int j; float rr;
  if (d < 256) { j = d; rr = r; }
  else {
    j = d - 256;
    const float l = log2f(r);
    rr = fminf(0.0025f * l * l, 1.f);
  }
  rr = fmaxf(rr, 1.f / 5000.f);
  int idx = (int)roundf(rr * 5000.f) - 1;
  idx = min(max(idx, 0), 4999);
  const float denom = powf(50.f, 2.f * (float)j / 256.f);
  const float arg = (float)idx / denom;
  const float pe = (j & 1) ? cosf(arg) : sinf(arg);
  x[(size_t)t * D_MODEL + d] = fused[(size_t)t * D_MODEL + d] + pe;
}

// ---------------------------------------------------------------------------
// Attention for one (batch, head): L=9, DH=64. Plain VALU + LDS (tiny).
// ---------------------------------------------------------------------------
__global__ __launch_bounds__(128) void attn_kernel(
    const float* __restrict__ qkv,    // [288][1536]
    float* __restrict__ o)            // [288][512]
{
  const int b = blockIdx.x >> 3, h = blockIdx.x & 7;
  __shared__ float q[9][64], k[9][64], v[9][64], sc[9][10];
  const int tid = threadIdx.x;
  for (int i = tid; i < 9 * 64; i += 128) {
    const int li = i / 64, d = i - li * 64;
    const float* row = qkv + (size_t)(b * 9 + li) * 1536;
    q[li][d] = row[h * 64 + d];
    k[li][d] = row[512 + h * 64 + d];
    v[li][d] = row[1024 + h * 64 + d];
  }
  __syncthreads();
  for (int p = tid; p < 81; p += 128) {
    const int qi = p / 9, ki = p - qi * 9;
    float s = 0.f;
#pragma unroll
    for (int d = 0; d < 64; ++d) s += q[qi][d] * k[ki][d];
    sc[qi][ki] = s * 0.125f;          // 1/sqrt(64)
  }
  __syncthreads();
  if (tid < 9) {
    float mx = -3.4e38f;
    for (int j = 0; j < 9; ++j) mx = fmaxf(mx, sc[tid][j]);
    float se = 0.f;
    for (int j = 0; j < 9; ++j) { const float e = expf(sc[tid][j] - mx); sc[tid][j] = e; se += e; }
    const float inv = 1.f / se;
    for (int j = 0; j < 9; ++j) sc[tid][j] *= inv;
  }
  __syncthreads();
  for (int p = tid; p < 9 * 64; p += 128) {
    const int qi = p / 64, d = p - qi * 64;
    float s = 0.f;
#pragma unroll
    for (int j = 0; j < 9; ++j) s += sc[qi][j] * v[j][d];
    o[(size_t)(b * 9 + qi) * D_MODEL + h * 64 + d] = s;
  }
}

// ---------------------------------------------------------------------------
// out = LayerNorm(x + y) * g + b, one block (256 threads) per token.
// ---------------------------------------------------------------------------
__global__ __launch_bounds__(256) void ln_residual_kernel(
    const float* __restrict__ x, const float* __restrict__ y,
    const float* __restrict__ g, const float* __restrict__ b,
    float* __restrict__ out)
{
  const int t = blockIdx.x, tid = threadIdx.x;
  __shared__ float red[256];
  __shared__ float mean_s, rstd_s;
  const float* xr = x + (size_t)t * D_MODEL;
  const float* yr = y + (size_t)t * D_MODEL;
  float vals[2], s = 0.f;
#pragma unroll
  for (int i = 0; i < 2; ++i) {
    const int d = tid + i * 256;
    vals[i] = xr[d] + yr[d];
    s += vals[i];
  }
  red[tid] = s; __syncthreads();
  for (int off = 128; off > 0; off >>= 1) {
    if (tid < off) red[tid] += red[tid + off];
    __syncthreads();
  }
  if (tid == 0) mean_s = red[0] * (1.f / 512.f);
  __syncthreads();
  const float m = mean_s;
  s = 0.f;
#pragma unroll
  for (int i = 0; i < 2; ++i) { const float d0 = vals[i] - m; s += d0 * d0; }
  red[tid] = s; __syncthreads();
  for (int off = 128; off > 0; off >>= 1) {
    if (tid < off) red[tid] += red[tid + off];
    __syncthreads();
  }
  if (tid == 0) rstd_s = rsqrtf(red[0] * (1.f / 512.f) + 1e-5f);
  __syncthreads();
  const float rs = rstd_s;
#pragma unroll
  for (int i = 0; i < 2; ++i) {
    const int d = tid + i * 256;
    out[(size_t)t * D_MODEL + d] = (vals[i] - m) * rs * g[d] + b[d];
  }
}

__global__ __launch_bounds__(512) void scale_kernel(
    const float* __restrict__ x, const float* __restrict__ frac,
    float* __restrict__ out)
{
  const int t = blockIdx.x, d = threadIdx.x;
  out[(size_t)t * D_MODEL + d] = x[(size_t)t * D_MODEL + d] * frac[t];
}

// ---------------------------------------------------------------------------
extern "C" void kernel_launch(void* const* d_in, const int* in_sizes, int n_in,
                              void* d_out, int out_size, void* d_ws, size_t ws_size,
                              hipStream_t stream)
{
  const int*   Z        = (const int*)d_in[0];
  const float* frac     = (const float*)d_in[1];
  const float* emb[3]   = {(const float*)d_in[2], (const float*)d_in[5], (const float*)d_in[8]};
  const float* projw[3] = {(const float*)d_in[3], (const float*)d_in[6], (const float*)d_in[9]};
  const float* projb[3] = {(const float*)d_in[4], (const float*)d_in[7], (const float*)d_in[10]};
  const float* keys     = (const float*)d_in[11];
  const float* router_w = (const float*)d_in[12];
  const float* router_b = (const float*)d_in[13];
  const float* exp_w1   = (const float*)d_in[14];
  const float* exp_b1   = (const float*)d_in[15];
  const float* exp_w2   = (const float*)d_in[16];
  const float* exp_b2   = (const float*)d_in[17];
  const float* qkv_w    = (const float*)d_in[18];
  const float* qkv_b    = (const float*)d_in[19];
  const float* ow       = (const float*)d_in[20];
  const float* ob       = (const float*)d_in[21];
  const float* ln1_g    = (const float*)d_in[22];
  const float* ln1_b    = (const float*)d_in[23];
  const float* ln2_g    = (const float*)d_in[24];
  const float* ln2_b    = (const float*)d_in[25];
  const float* fw1      = (const float*)d_in[26];
  const float* fb1      = (const float*)d_in[27];
  const float* fw2      = (const float*)d_in[28];
  const float* fb2      = (const float*)d_in[29];

  float* ws    = (float*)d_ws;
  float* views = ws;                                       // 3*288*512
  float* wfull = views + (size_t)N_VIEW * T_TOK * D_MODEL; // 3*288*16
  float* fused = wfull + (size_t)N_VIEW * T_TOK * N_EXP;   // 288*512
  float* x     = fused + (size_t)T_TOK * D_MODEL;          // 288*512
  float* qkv   = x     + (size_t)T_TOK * D_MODEL;          // 288*1536
  float* obuf  = qkv   + (size_t)T_TOK * 1536;             // 288*512
  float* tmp   = obuf  + (size_t)T_TOK * D_MODEL;          // 288*512
  float* hid   = tmp   + (size_t)T_TOK * D_MODEL;          // 288*2048

  const int featK[3] = {200, 132, 112};

  // 1) per-view embedding gather + projection to d_model
  for (int vi = 0; vi < 3; ++vi) {
    const int tiles = (T_TOK / 16) * (D_MODEL / 16);
    gemm_wmma_kernel<0, true><<<(tiles + 7) / 8, 256, 0, stream>>>(
        emb[vi], projw[vi], projb[vi], views + (size_t)vi * T_TOK * D_MODEL, Z,
        T_TOK, D_MODEL, featK[vi], featK[vi], D_MODEL, D_MODEL);
  }

  // 2) Laplace gating -> dense top-4 weights
  gating_kernel<<<N_VIEW * T_TOK, 256, 0, stream>>>(views, keys, router_w,
                                                    router_b, wfull);

  // 3) dense expert MLPs, gate-weighted accumulation
  zero_kernel<<<(T_TOK * D_MODEL + 255) / 256, 256, 0, stream>>>(fused,
                                                                 T_TOK * D_MODEL);
  expert_kernel<<<dim3(T_TOK / 16, N_EXP, N_VIEW), 512, 0, stream>>>(
      views, wfull, exp_w1, exp_b1, exp_w2, exp_b2, fused);

  // 4) fractional positional encoding
  pe_add_kernel<<<T_TOK, 512, 0, stream>>>(fused, frac, x);

  // 5) transformer stack
  for (int i = 0; i < 3; ++i) {
    {
      const int tiles = (T_TOK / 16) * (1536 / 16);
      gemm_wmma_kernel<0, false><<<(tiles + 7) / 8, 256, 0, stream>>>(
          x, qkv_w + (size_t)i * 512 * 1536, qkv_b + (size_t)i * 1536, qkv,
          nullptr, T_TOK, 1536, 512, 512, 1536, 1536);
    }
    attn_kernel<<<32 * 8, 128, 0, stream>>>(qkv, obuf);
    {
      const int tiles = (T_TOK / 16) * (512 / 16);
      gemm_wmma_kernel<0, false><<<(tiles + 7) / 8, 256, 0, stream>>>(
          obuf, ow + (size_t)i * 512 * 512, ob + (size_t)i * 512, tmp,
          nullptr, T_TOK, 512, 512, 512, 512, 512);
    }
    ln_residual_kernel<<<T_TOK, 256, 0, stream>>>(
        x, tmp, ln1_g + (size_t)i * 512, ln1_b + (size_t)i * 512, x);
    {
      const int tiles = (T_TOK / 16) * (2048 / 16);
      gemm_wmma_kernel<2, false><<<(tiles + 7) / 8, 256, 0, stream>>>(
          x, fw1 + (size_t)i * 512 * 2048, fb1 + (size_t)i * 2048, hid,
          nullptr, T_TOK, 2048, 512, 512, 2048, 2048);
    }
    {
      const int tiles = (T_TOK / 16) * (512 / 16);
      gemm_wmma_kernel<0, false><<<(tiles + 7) / 8, 256, 0, stream>>>(
          hid, fw2 + (size_t)i * 2048 * 512, fb2 + (size_t)i * 512, tmp,
          nullptr, T_TOK, 512, 2048, 2048, 512, 512);
    }
    ln_residual_kernel<<<T_TOK, 256, 0, stream>>>(
        x, tmp, ln2_g + (size_t)i * 512, ln2_b + (size_t)i * 512, x);
  }

  // 6) final scale by frac -> output
  scale_kernel<<<T_TOK, 512, 0, stream>>>(x, frac, (float*)d_out);

  (void)in_sizes; (void)n_in; (void)out_size; (void)ws_size;
}
